// MultiHeadAttn_21784074125899
// MI455X (gfx1250) — matrix operand order repaired and checked by reading, not compiled, not run on hardware
//
#include <hip/hip_runtime.h>

// ---------------------------------------------------------------------------
// MI455X (gfx1250) fused 1x1-conv QKV projection + row attention + residual.
// Split-bf16 (hi+lo) WMMA everywhere: D += Ah*Bh + Ah*Bl + Al*Bh, fp32 acc.
// One workgroup (8 wave32) per (b,h) slice; 1024 workgroups.
// ---------------------------------------------------------------------------

typedef __bf16 bf16_t;
typedef __bf16 v8bf  __attribute__((ext_vector_type(8)));
typedef __bf16 v16bf __attribute__((ext_vector_type(16)));
typedef float  v8f   __attribute__((ext_vector_type(8)));

#define C_DIM 256
#define W_DIM 64
#define H_DIM 64
#define HW    4096          // H*W, channel stride in NCHW
#define RS    264           // padded row stride (bf16 elems) for [64][256] mats
#define RSS   72            // padded row stride for [64][64] S and [256][64] Vt

__device__ __forceinline__ v16bf frag_cat(v8bf a, v8bf b) {
    return __builtin_shufflevector(a, b, 0,1,2,3,4,5,6,7,8,9,10,11,12,13,14,15);
}

__device__ __forceinline__ v8f wmma_bf16(v16bf a, v16bf b, v8f c) {
    // v_wmma_f32_16x16x32_bf16  (8-arg form, probe-confirmed)
    return __builtin_amdgcn_wmma_f32_16x16x32_bf16(false, a, false, b,
                                                   (short)0, c, false, false);
}

__device__ __forceinline__ void split_bf16(float v, bf16_t& h, bf16_t& l) {
    h = (bf16_t)v;
    l = (bf16_t)(v - (float)h);
}

// Prologue: convert wq/wk/wv fp32 [256][256] -> bf16 hi/lo planes in d_ws.
// ws layout (bf16 elems): [q_hi 65536][q_lo 65536][k_hi][k_lo][v_hi][v_lo]
__global__ __launch_bounds__(256) void cvt_weights_kernel(
    const float* __restrict__ wq, const float* __restrict__ wk,
    const float* __restrict__ wv, bf16_t* __restrict__ dst)
{
    int idx = blockIdx.x * 256 + threadIdx.x;      // 0 .. 3*65536-1
    int plane = idx >> 16;
    int j = idx & 65535;
    const float* src = (plane == 0) ? wq : (plane == 1) ? wk : wv;
    bf16_t h, l;
    split_bf16(src[j], h, l);
    dst[(size_t)plane * 131072 + j]         = h;
    dst[(size_t)plane * 131072 + 65536 + j] = l;
}

__global__ __launch_bounds__(256) void attn_fused_kernel(
    const float* __restrict__ a,
    const bf16_t* __restrict__ wsb,     // converted weights (6 planes)
    const float* __restrict__ bq,
    const float* __restrict__ bk,
    const float* __restrict__ bv,
    float* __restrict__ out)
{
    // ---- LDS regions (270 KB total, <= 320 KB/WGP) ----
    // [0      , 67584) : Ahi [64][264] + Alo      (Xt, bf16) ; reused for S hi/lo
    // [67584  ,135168) : Qhi + Qlo                ; reused for O fp32 [64][264]
    // [135168 ,202752) : Khi + Klo
    // [202752 ,276480) : Vt hi [256][72] + Vt lo  (V transposed)
    __shared__ __attribute__((aligned(16))) unsigned char smem[276480];
    bf16_t* Ahi = (bf16_t*)smem;
    bf16_t* Alo = Ahi + 64 * RS;
    bf16_t* Qhi = (bf16_t*)(smem + 67584);
    bf16_t* Qlo = Qhi + 64 * RS;
    bf16_t* Khi = (bf16_t*)(smem + 135168);
    bf16_t* Klo = Khi + 64 * RS;
    bf16_t* Vhi = (bf16_t*)(smem + 202752);
    bf16_t* Vlo = Vhi + 256 * RSS;
    bf16_t* Shi = (bf16_t*)smem;                 // [64][72] overlays A (after sync)
    bf16_t* Slo = Shi + 64 * RSS;
    float*  Of  = (float*)(smem + 67584);        // [64][264] overlays Q (after sync)

    const int t    = threadIdx.x;
    const int lane = t & 31;
    const int wid  = t >> 5;        // 0..7
    const int lm   = lane & 15;     // tile column (n) / tile row (m)
    const int lh   = lane >> 4;     // half-wave selector

    const int bb = blockIdx.x >> 6;     // batch
    const int hh = blockIdx.x & 63;     // row h
    const float* Xg = a + (size_t)bb * C_DIM * HW + (size_t)hh * W_DIM; // [c][w], c-stride HW

    // ---------------- Phase 0: stage Xt = a[b,:,h,:]^T into LDS (hi/lo bf16) --
    for (int i = 0; i < 64; ++i) {
        int idx = i * 256 + t;
        int c = idx >> 6, w = idx & 63;          // coalesced along w
        float x = Xg[(size_t)c * HW + w];
        bf16_t h, l; split_bf16(x, h, l);
        Ahi[w * RS + c] = h;
        Alo[w * RS + c] = l;
    }
    __syncthreads();

    // ---------------- Phase 1: Q/K/V = Xt * W^T + b  (64x256, K=256) ---------
    // 6 units per wave: unit u -> proj p = u/2, otile = wid*2 + (u&1).
    // B fragments read weight rows once per WG; A fragments re-read from LDS.
    for (int u = 0; u < 6; ++u) {
        const int p     = u >> 1;
        const int ot    = wid * 2 + (u & 1);
        const int obase = ot * 16;
        const bf16_t* Whi = wsb + (size_t)p * 131072;
        const bf16_t* Wlo = Whi + 65536;
        const float*  bp  = (p == 0) ? bq : (p == 1) ? bk : bv;

        const float bias = bp[obase + lm];       // D col n = lane%16
        v8f acc[4];
        #pragma unroll
        for (int wt = 0; wt < 4; ++wt)
            #pragma unroll
            for (int e = 0; e < 8; ++e) acc[wt][e] = bias;

        for (int ks = 0; ks < 8; ++ks) {
            // B frag (32x16): lane -> col o = obase+lm, 16 contiguous k at lh*16
            const bf16_t* wh = Whi + (size_t)(obase + lm) * C_DIM + ks * 32 + lh * 16;
            const bf16_t* wl = Wlo + (size_t)(obase + lm) * C_DIM + ks * 32 + lh * 16;
            v16bf Bh = frag_cat(*(const v8bf*)wh, *(const v8bf*)(wh + 8));
            v16bf Bl = frag_cat(*(const v8bf*)wl, *(const v8bf*)(wl + 8));
            #pragma unroll
            for (int wt = 0; wt < 4; ++wt) {
                // A frag (16x32): lane -> row w, 16 contiguous k at lh*16
                const bf16_t* ah = Ahi + (wt * 16 + lm) * RS + ks * 32 + lh * 16;
                const bf16_t* al = Alo + (wt * 16 + lm) * RS + ks * 32 + lh * 16;
                v16bf Ah = frag_cat(*(const v8bf*)ah, *(const v8bf*)(ah + 8));
                v16bf Al = frag_cat(*(const v8bf*)al, *(const v8bf*)(al + 8));
                acc[wt] = wmma_bf16(Ah, Bh, acc[wt]);
                acc[wt] = wmma_bf16(Ah, Bl, acc[wt]);
                acc[wt] = wmma_bf16(Al, Bh, acc[wt]);
            }
        }
        // Store results (hi/lo bf16). Q,K row-major [w][o]; V transposed [o][w].
        #pragma unroll
        for (int wt = 0; wt < 4; ++wt) {
            #pragma unroll
            for (int r = 0; r < 8; ++r) {
                int w = wt * 16 + r + 8 * lh;            // D row m = r (+8 hi half)
                bf16_t hv, lv; split_bf16(acc[wt][r], hv, lv);
                if (p == 0)      { Qhi[w * RS + obase + lm] = hv; Qlo[w * RS + obase + lm] = lv; }
                else if (p == 1) { Khi[w * RS + obase + lm] = hv; Klo[w * RS + obase + lm] = lv; }
                else             { Vhi[(obase + lm) * RSS + w] = hv; Vlo[(obase + lm) * RSS + w] = lv; }
            }
        }
    }
    __syncthreads();

    // ---------------- Phase 2: S = Q * K^T  (64x64, K=256) -------------------
    {
        const int wt  = wid >> 1;                 // both tiles share wt
        const int xtb = (wid & 1) * 2;            // xt = xtb, xtb+1
        v8f acc0 = {}, acc1 = {};
        for (int ks = 0; ks < 8; ++ks) {
            const bf16_t* qh = Qhi + (wt * 16 + lm) * RS + ks * 32 + lh * 16;
            const bf16_t* ql = Qlo + (wt * 16 + lm) * RS + ks * 32 + lh * 16;
            v16bf QhF = frag_cat(*(const v8bf*)qh, *(const v8bf*)(qh + 8));
            v16bf QlF = frag_cat(*(const v8bf*)ql, *(const v8bf*)(ql + 8));
            const bf16_t* k0h = Khi + ((xtb    ) * 16 + lm) * RS + ks * 32 + lh * 16;
            const bf16_t* k0l = Klo + ((xtb    ) * 16 + lm) * RS + ks * 32 + lh * 16;
            const bf16_t* k1h = Khi + ((xtb + 1) * 16 + lm) * RS + ks * 32 + lh * 16;
            const bf16_t* k1l = Klo + ((xtb + 1) * 16 + lm) * RS + ks * 32 + lh * 16;
            v16bf K0h = frag_cat(*(const v8bf*)k0h, *(const v8bf*)(k0h + 8));
            v16bf K0l = frag_cat(*(const v8bf*)k0l, *(const v8bf*)(k0l + 8));
            v16bf K1h = frag_cat(*(const v8bf*)k1h, *(const v8bf*)(k1h + 8));
            v16bf K1l = frag_cat(*(const v8bf*)k1l, *(const v8bf*)(k1l + 8));
            acc0 = wmma_bf16(QhF, K0h, acc0);
            acc0 = wmma_bf16(QhF, K0l, acc0);
            acc0 = wmma_bf16(QlF, K0h, acc0);
            acc1 = wmma_bf16(QhF, K1h, acc1);
            acc1 = wmma_bf16(QhF, K1l, acc1);
            acc1 = wmma_bf16(QlF, K1h, acc1);
        }
        #pragma unroll
        for (int r = 0; r < 8; ++r) {
            int w = wt * 16 + r + 8 * lh;
            bf16_t hv, lv;
            split_bf16(acc0[r], hv, lv);
            Shi[w * RSS + xtb * 16 + lm] = hv;  Slo[w * RSS + xtb * 16 + lm] = lv;
            split_bf16(acc1[r], hv, lv);
            Shi[w * RSS + (xtb + 1) * 16 + lm] = hv;  Slo[w * RSS + (xtb + 1) * 16 + lm] = lv;
        }
    }
    __syncthreads();

    // ---------------- Phase 3: O = S * V  (64x256, K=64) ---------------------
    for (int s = 0; s < 2; ++s) {
        const int obase = (wid * 2 + s) * 16;
        v8f acc[4] = {v8f{}, v8f{}, v8f{}, v8f{}};
        #pragma unroll
        for (int xs = 0; xs < 2; ++xs) {
            // B frag from Vt: lane -> col o, 16 contiguous x at lh*16
            const bf16_t* vh = Vhi + (obase + lm) * RSS + xs * 32 + lh * 16;
            const bf16_t* vl = Vlo + (obase + lm) * RSS + xs * 32 + lh * 16;
            v16bf Bh = frag_cat(*(const v8bf*)vh, *(const v8bf*)(vh + 8));
            v16bf Bl = frag_cat(*(const v8bf*)vl, *(const v8bf*)(vl + 8));
            #pragma unroll
            for (int wt = 0; wt < 4; ++wt) {
                const bf16_t* sh = Shi + (wt * 16 + lm) * RSS + xs * 32 + lh * 16;
                const bf16_t* sl = Slo + (wt * 16 + lm) * RSS + xs * 32 + lh * 16;
                v16bf Ah = frag_cat(*(const v8bf*)sh, *(const v8bf*)(sh + 8));
                v16bf Al = frag_cat(*(const v8bf*)sl, *(const v8bf*)(sl + 8));
                acc[wt] = wmma_bf16(Ah, Bh, acc[wt]);
                acc[wt] = wmma_bf16(Ah, Bl, acc[wt]);
                acc[wt] = wmma_bf16(Al, Bh, acc[wt]);
            }
        }
        #pragma unroll
        for (int wt = 0; wt < 4; ++wt)
            #pragma unroll
            for (int r = 0; r < 8; ++r) {
                int w = wt * 16 + r + 8 * lh;
                Of[w * RS + obase + lm] = acc[wt][r];
            }
    }
    __syncthreads();

    // ---------------- Phase 4: out = a + O^T (coalesced) ---------------------
    float* Og = out + (size_t)bb * C_DIM * HW + (size_t)hh * W_DIM;
    for (int i = 0; i < 64; ++i) {
        int idx = i * 256 + t;
        int w = idx & 63, o = idx >> 6;
        Og[(size_t)o * HW + w] = Xg[(size_t)o * HW + w] + Of[w * RS + o];
    }
}

extern "C" void kernel_launch(void* const* d_in, const int* in_sizes, int n_in,
                              void* d_out, int out_size, void* d_ws, size_t ws_size,
                              hipStream_t stream) {
    (void)in_sizes; (void)n_in; (void)out_size; (void)ws_size;
    const float* a  = (const float*)d_in[0];
    const float* wq = (const float*)d_in[1];
    const float* bq = (const float*)d_in[2];
    const float* wk = (const float*)d_in[3];
    const float* bk = (const float*)d_in[4];
    const float* wv = (const float*)d_in[5];
    const float* bv = (const float*)d_in[6];
    bf16_t* wsb = (bf16_t*)d_ws;     // needs 786432 bytes

    cvt_weights_kernel<<<768, 256, 0, stream>>>(wq, wk, wv, wsb);
    attn_fused_kernel<<<1024, 256, 0, stream>>>(a, wsb, bq, bk, bv, (float*)d_out);
}
